// C_MFN_35794257445299
// MI455X (gfx1250) — matrix operand order repaired and checked by reading, not compiled, not run on hardware
//
#include <hip/hip_runtime.h>
#include <hip/hip_bf16.h>
#include <utility>

// ---------------- types ----------------
typedef __attribute__((ext_vector_type(16))) _Float16 v16h;
typedef __attribute__((ext_vector_type(8)))  float    v8f;

#define BATCH_N 4096
#define XP_LD   752
#define SEQ_T   20

union FragU { v16h v; uint4 q[2]; };

// ---------------- fast activations (branch-free, v_exp_f32 based) -----------
__device__ __forceinline__ float fast_sigmoid(float x) {
  return 1.f / (1.f + __expf(-x));
}
__device__ __forceinline__ float fast_tanh(float x) {
  float xc = fminf(fmaxf(x, -15.f), 15.f);
  float e  = __expf(2.f * xc);
  return (e - 1.f) / (e + 1.f);
}

// ---------------- WMMA fragment loads from precomputed lane pointers --------
// A 16x32 (MxK): lanes 0-15 row M=lane, K {k0..k0+7, k0+16..k0+23};
//                lanes 16-31 row M=lane-16, K {k0+8..k0+15, k0+24..k0+31}.
// (lane base pointer already includes the +8 half-offset for lanes 16-31)
__device__ __forceinline__ v16h frag_a(const _Float16* __restrict__ p, int k0) {
  FragU f;
  f.q[0] = *(const uint4*)(p + k0);
  f.q[1] = *(const uint4*)(p + k0 + 16);
  return f.v;
}
// B 32x16 (KxN) from W[N,K] row-major: lane n<16 -> column n, K {k0..k0+7, k0+16..k0+23};
// lanes 16-31 -> column n-16, K {k0+8..k0+15, k0+24..k0+31} (base includes +8 offset).
__device__ __forceinline__ v16h frag_b(const _Float16* __restrict__ p, int k0) {
  FragU f;
  f.q[0] = *(const uint4*)(p + k0);
  f.q[1] = *(const uint4*)(p + k0 + 16);
  return f.v;
}

// ---------------- fused WMMA GEMM: C = act(A[M,K] * W[N,K]^T + bias) ----------
// grid = (N/128, M/64), block = 128 (4 waves); wave tile = 32(M) x 64(N).
// K is a template constant: the K-loop is fully unrolled, so every fragment
// load becomes global_load_b128 with an immediate offset (no per-step address
// VALU, no double-buffer copies); the scheduler pipelines loads across WMMAs.
template <int K, int ACT, bool O32, bool O16>
__global__ void __launch_bounds__(128) wmma_gemm_kernel(
    const _Float16* __restrict__ A, int lda,
    const _Float16* __restrict__ W, int ldw,
    const float* __restrict__ bias,
    float* __restrict__ o32, _Float16* __restrict__ o16,
    int N)
{
  const int lane = threadIdx.x & 31;
  const int wave = threadIdx.x >> 5;
  const int m0 = blockIdx.y * 64 + (wave & 1) * 32;
  const int n0 = blockIdx.x * 128 + (wave >> 1) * 64;

  const int lrow = lane & 15;
  const int hi   = (lane >> 4) & 1;

  // per-lane base pointers (computed once)
  const _Float16* pa0 = A + (size_t)(m0 + lrow) * lda + hi * 8;
  const _Float16* pa1 = pa0 + (size_t)16 * lda;
  const _Float16* pb0 = W + (size_t)(n0 + lrow) * ldw + hi * 8;
  const _Float16* pb1 = pb0 + (size_t)16 * ldw;
  const _Float16* pb2 = pb1 + (size_t)16 * ldw;
  const _Float16* pb3 = pb2 + (size_t)16 * ldw;

  v8f zero = {};
  v8f acc[2][4];
#pragma unroll
  for (int i = 0; i < 2; ++i)
#pragma unroll
    for (int j = 0; j < 4; ++j) acc[i][j] = zero;

#pragma unroll
  for (int k0 = 0; k0 < K; k0 += 32) {
    v16h a0 = frag_a(pa0, k0);
    v16h a1 = frag_a(pa1, k0);
    v16h b0 = frag_b(pb0, k0);
    v16h b1 = frag_b(pb1, k0);
    v16h b2 = frag_b(pb2, k0);
    v16h b3 = frag_b(pb3, k0);
    acc[0][0] = __builtin_amdgcn_wmma_f32_16x16x32_f16(false, a0, false, b0, (short)0, acc[0][0], false, false);
    acc[1][0] = __builtin_amdgcn_wmma_f32_16x16x32_f16(false, a1, false, b0, (short)0, acc[1][0], false, false);
    acc[0][1] = __builtin_amdgcn_wmma_f32_16x16x32_f16(false, a0, false, b1, (short)0, acc[0][1], false, false);
    acc[1][1] = __builtin_amdgcn_wmma_f32_16x16x32_f16(false, a1, false, b1, (short)0, acc[1][1], false, false);
    acc[0][2] = __builtin_amdgcn_wmma_f32_16x16x32_f16(false, a0, false, b2, (short)0, acc[0][2], false, false);
    acc[1][2] = __builtin_amdgcn_wmma_f32_16x16x32_f16(false, a1, false, b2, (short)0, acc[1][2], false, false);
    acc[0][3] = __builtin_amdgcn_wmma_f32_16x16x32_f16(false, a0, false, b3, (short)0, acc[0][3], false, false);
    acc[1][3] = __builtin_amdgcn_wmma_f32_16x16x32_f16(false, a1, false, b3, (short)0, acc[1][3], false, false);
  }

  // C/D 16x16 f32 layout: VGPR r -> lanes0-15 (M=r,N=lane), lanes16-31 (M=r+8,N=lane-16)
  const int col = lane & 15;
  const int rb  = hi * 8;
#pragma unroll
  for (int i = 0; i < 2; ++i) {
#pragma unroll
    for (int j = 0; j < 4; ++j) {
      const int nc = n0 + j * 16 + col;
      const float bv = bias[nc];
#pragma unroll
      for (int r = 0; r < 8; ++r) {
        const int row = m0 + i * 16 + rb + r;
        float v = acc[i][j][r] + bv;
        if (ACT == 1) v = fmaxf(v, 0.f);
        else if (ACT == 2) v = fast_sigmoid(v);
        else if (ACT == 3) v = fast_tanh(v);
        if (O32) o32[(size_t)row * N + nc] = v;
        if (O16) o16[(size_t)row * N + nc] = (_Float16)v;
      }
    }
  }
}

// ---------------- elementwise / pack kernels ----------------
__global__ void k_convert_f16(const float* __restrict__ in, _Float16* __restrict__ out, int n) {
  int i = blockIdx.x * blockDim.x + threadIdx.x;
  if (i < n) out[i] = (_Float16)in[i];
}

__global__ void k_zero_f32(float* __restrict__ p, int n) {
  int i = blockIdx.x * blockDim.x + threadIdx.x;
  if (i < n) p[i] = 0.f;
}

// fused [Wih | Whh | 0pad] -> f16 [N, kp]
__global__ void k_pack_lstm_w(const float* __restrict__ Wih, const float* __restrict__ Whh,
                              _Float16* __restrict__ out, int N, int inlen, int hlen, int kp) {
  int i = blockIdx.x * blockDim.x + threadIdx.x;
  if (i >= N * kp) return;
  int n = i / kp, c = i % kp;
  float v = 0.f;
  if (c < inlen)             v = Wih[(size_t)n * inlen + c];
  else if (c < inlen + hlen) v = Whh[(size_t)n * hlen + (c - inlen)];
  out[i] = (_Float16)v;
}

// fused [x_t slice | h | 0pad] -> f16 [BATCH, kp]
__global__ void k_pack_xh(const float* __restrict__ x, int x_off, int inlen,
                          const float* __restrict__ h, int hlen,
                          _Float16* __restrict__ out, int kp) {
  int i = blockIdx.x * blockDim.x + threadIdx.x;
  if (i >= BATCH_N * kp) return;
  int row = i / kp, c = i % kp;
  float v = 0.f;
  if (c < inlen)             v = x[(size_t)row * XP_LD + x_off + c];
  else if (c < inlen + hlen) v = h[(size_t)row * hlen + (c - inlen)];
  out[i] = (_Float16)v;
}

// z[B,4H] (i,f,g,o) + c_prev -> c_new, h_new
__global__ void k_lstm_pointwise(const float* __restrict__ z, const float* __restrict__ c_prev,
                                 float* __restrict__ c_new, float* __restrict__ h_new, int H) {
  int i = blockIdx.x * blockDim.x + threadIdx.x;
  if (i >= BATCH_N * H) return;
  int row = i / H, col = i % H;
  const float* zr = z + (size_t)row * 4 * H;
  float iv = fast_sigmoid(zr[col]);
  float fv = fast_sigmoid(zr[H + col]);
  float gv = fast_tanh(zr[2 * H + col]);
  float ov = fast_sigmoid(zr[3 * H + col]);
  float c2 = fv * c_prev[i] + iv * gv;
  c_new[i] = c2;
  h_new[i] = ov * fast_tanh(c2);
}

// c_star = [t_c_prev|a_c_prev|v_c_prev|t_c|a_c|v_c] -> f16 [B,512]
__global__ void k_pack_cstar(const float* __restrict__ tcp, const float* __restrict__ acp,
                             const float* __restrict__ vcp, const float* __restrict__ tc,
                             const float* __restrict__ ac,  const float* __restrict__ vc,
                             _Float16* __restrict__ out) {
  int i = blockIdx.x * blockDim.x + threadIdx.x;
  if (i >= BATCH_N * 512) return;
  int row = i >> 9, c = i & 511;
  float v;
  if      (c < 128) v = tcp[(size_t)row * 128 + c];
  else if (c < 192) v = acp[(size_t)row * 64 + (c - 128)];
  else if (c < 256) v = vcp[(size_t)row * 64 + (c - 192)];
  else if (c < 384) v = tc [(size_t)row * 128 + (c - 256)];
  else if (c < 448) v = ac [(size_t)row * 64 + (c - 384)];
  else              v = vc [(size_t)row * 64 + (c - 448)];
  out[i] = (_Float16)v;
}

// softmax(scores[row,:512]) * c_star -> both[row, 0:512] (f16). One wave per row.
__global__ void __launch_bounds__(256) k_softmax_attend(
    const float* __restrict__ scores, const _Float16* __restrict__ cstar,
    _Float16* __restrict__ both) {
  int lane = threadIdx.x & 31;
  int row  = blockIdx.x * 8 + (threadIdx.x >> 5);
  const float* s = scores + (size_t)row * 512;
  float v[16];
  float mx = -1e30f;
#pragma unroll
  for (int i = 0; i < 16; ++i) { v[i] = s[lane + i * 32]; mx = fmaxf(mx, v[i]); }
  for (int o = 16; o >= 1; o >>= 1) mx = fmaxf(mx, __shfl_xor(mx, o, 32));
  float sum = 0.f;
#pragma unroll
  for (int i = 0; i < 16; ++i) { v[i] = __expf(v[i] - mx); sum += v[i]; }
  for (int o = 16; o >= 1; o >>= 1) sum += __shfl_xor(sum, o, 32);
  float inv = 1.f / sum;
  _Float16* brow = both + (size_t)row * 768;
  const _Float16* cr = cstar + (size_t)row * 512;
#pragma unroll
  for (int i = 0; i < 16; ++i) {
    int c = lane + i * 32;
    brow[c] = (_Float16)(v[i] * inv * (float)cr[c]);
  }
}

// mem (pre-update) -> both[:, 512:768] f16
__global__ void k_pack_mem(const float* __restrict__ mem, _Float16* __restrict__ both) {
  int i = blockIdx.x * blockDim.x + threadIdx.x;
  if (i >= BATCH_N * 256) return;
  int row = i >> 8, c = i & 255;
  both[(size_t)row * 768 + 512 + c] = (_Float16)mem[i];
}

// mem = gamma1*mem + gamma2*c_hat   (all f32 [B,256], in-place on mem)
__global__ void k_mem_update(float* __restrict__ mem, const float* __restrict__ g1,
                             const float* __restrict__ g2, const float* __restrict__ chat) {
  int i = blockIdx.x * blockDim.x + threadIdx.x;
  if (i >= BATCH_N * 256) return;
  mem[i] = g1[i] * mem[i] + g2[i] * chat[i];
}

// last_h = [t_h|a_h|v_h|mem] -> f16 [B,512]
__global__ void k_pack_lasth(const float* __restrict__ th, const float* __restrict__ ah,
                             const float* __restrict__ vh, const float* __restrict__ mem,
                             _Float16* __restrict__ out) {
  int i = blockIdx.x * blockDim.x + threadIdx.x;
  if (i >= BATCH_N * 512) return;
  int row = i >> 9, c = i & 511;
  float v;
  if      (c < 128) v = th[(size_t)row * 128 + c];
  else if (c < 192) v = ah[(size_t)row * 64 + (c - 128)];
  else if (c < 256) v = vh[(size_t)row * 64 + (c - 192)];
  else              v = mem[(size_t)row * 256 + (c - 256)];
  out[i] = (_Float16)v;
}

// out[row] = hrelu[row,:128] . w2 + b2   (final GEMV, N=1)
__global__ void k_final(const _Float16* __restrict__ h, const _Float16* __restrict__ w2,
                        const float* __restrict__ b2, float* __restrict__ out) {
  int row = blockIdx.x * blockDim.x + threadIdx.x;
  if (row >= BATCH_N) return;
  const _Float16* hr = h + (size_t)row * 128;
  float acc = b2[0];
#pragma unroll 8
  for (int k = 0; k < 128; ++k) acc += (float)hr[k] * (float)w2[k];
  out[row] = acc;
}

// ---------------- host orchestration ----------------
static inline dim3 ew_grid(int n) { return dim3((n + 255) / 256); }

template <int K, int ACT, bool O32, bool O16>
static inline void gemm_launch(hipStream_t stream,
                               const _Float16* A, int lda, const _Float16* W, int ldw,
                               const float* bias, float* o32, _Float16* o16, int N) {
  dim3 grid(N / 128, BATCH_N / 64);
  wmma_gemm_kernel<K, ACT, O32, O16><<<grid, 128, 0, stream>>>(A, lda, W, ldw, bias, o32, o16, N);
}

extern "C" void kernel_launch(void* const* d_in, const int* in_sizes, int n_in,
                              void* d_out, int out_size, void* d_ws, size_t ws_size,
                              hipStream_t stream) {
  (void)in_sizes; (void)n_in; (void)out_size; (void)ws_size;
  const float* x_p   = (const float*)d_in[0];
  const float* c_t0  = (const float*)d_in[1];
  const float* c_a0  = (const float*)d_in[2];
  const float* c_v0  = (const float*)d_in[3];
  const float* mem0  = (const float*)d_in[4];
  const float* t_Wih = (const float*)d_in[5];
  const float* t_Whh = (const float*)d_in[6];
  const float* t_b   = (const float*)d_in[7];
  const float* a_Wih = (const float*)d_in[8];
  const float* a_Whh = (const float*)d_in[9];
  const float* a_b   = (const float*)d_in[10];
  const float* v_Wih = (const float*)d_in[11];
  const float* v_Whh = (const float*)d_in[12];
  const float* v_b   = (const float*)d_in[13];
  const float* a1W1 = (const float*)d_in[14]; const float* a1b1 = (const float*)d_in[15];
  const float* a1W2 = (const float*)d_in[16]; const float* a1b2 = (const float*)d_in[17];
  const float* a2W1 = (const float*)d_in[18]; const float* a2b1 = (const float*)d_in[19];
  const float* a2W2 = (const float*)d_in[20]; const float* a2b2 = (const float*)d_in[21];
  const float* g1W1 = (const float*)d_in[22]; const float* g1b1 = (const float*)d_in[23];
  const float* g1W2 = (const float*)d_in[24]; const float* g1b2 = (const float*)d_in[25];
  const float* g2W1 = (const float*)d_in[26]; const float* g2b1 = (const float*)d_in[27];
  const float* g2W2 = (const float*)d_in[28]; const float* g2b2 = (const float*)d_in[29];
  const float* oW1  = (const float*)d_in[30]; const float* ob1  = (const float*)d_in[31];
  const float* oW2  = (const float*)d_in[32]; const float* ob2  = (const float*)d_in[33];

  // ---- workspace bump allocator (256B aligned) ----
  char* base = (char*)d_ws; size_t off = 0;
  auto alloc = [&](size_t bytes) -> void* {
    void* r = base + off; off += (bytes + 255) & ~(size_t)255; return r;
  };
  // packed / converted f16 weights
  _Float16* wT   = (_Float16*)alloc(512 * 448 * 2);
  _Float16* wA   = (_Float16*)alloc(256 * 160 * 2);
  _Float16* wV   = (_Float16*)alloc(256 * 448 * 2);
  _Float16* fa1W1 = (_Float16*)alloc(128 * 512 * 2);
  _Float16* fa1W2 = (_Float16*)alloc(512 * 128 * 2);
  _Float16* fa2W1 = (_Float16*)alloc(128 * 512 * 2);
  _Float16* fa2W2 = (_Float16*)alloc(256 * 128 * 2);
  _Float16* fg1W1 = (_Float16*)alloc(128 * 768 * 2);
  _Float16* fg1W2 = (_Float16*)alloc(256 * 128 * 2);
  _Float16* fg2W1 = (_Float16*)alloc(128 * 768 * 2);
  _Float16* fg2W2 = (_Float16*)alloc(256 * 128 * 2);
  _Float16* foW1  = (_Float16*)alloc(128 * 512 * 2);
  _Float16* foW2  = (_Float16*)alloc(128 * 2);
  // activations
  _Float16* xhT = (_Float16*)alloc((size_t)BATCH_N * 448 * 2);
  _Float16* xhA = (_Float16*)alloc((size_t)BATCH_N * 160 * 2);
  _Float16* xhV = (_Float16*)alloc((size_t)BATCH_N * 448 * 2);
  float* zT = (float*)alloc((size_t)BATCH_N * 512 * 4);
  float* zA = (float*)alloc((size_t)BATCH_N * 256 * 4);
  float* zV = (float*)alloc((size_t)BATCH_N * 256 * 4);
  float* cT[2] = { (float*)alloc((size_t)BATCH_N * 128 * 4), (float*)alloc((size_t)BATCH_N * 128 * 4) };
  float* cA[2] = { (float*)alloc((size_t)BATCH_N * 64 * 4),  (float*)alloc((size_t)BATCH_N * 64 * 4) };
  float* cV[2] = { (float*)alloc((size_t)BATCH_N * 64 * 4),  (float*)alloc((size_t)BATCH_N * 64 * 4) };
  float* hT[2] = { (float*)alloc((size_t)BATCH_N * 128 * 4), (float*)alloc((size_t)BATCH_N * 128 * 4) };
  float* hA[2] = { (float*)alloc((size_t)BATCH_N * 64 * 4),  (float*)alloc((size_t)BATCH_N * 64 * 4) };
  float* hV[2] = { (float*)alloc((size_t)BATCH_N * 64 * 4),  (float*)alloc((size_t)BATCH_N * 64 * 4) };
  float*    memb  = (float*)alloc((size_t)BATCH_N * 256 * 4);
  _Float16* cstar = (_Float16*)alloc((size_t)BATCH_N * 512 * 2);
  _Float16* h1    = (_Float16*)alloc((size_t)BATCH_N * 128 * 2);
  float*    scores = (float*)alloc((size_t)BATCH_N * 512 * 4);
  _Float16* both  = (_Float16*)alloc((size_t)BATCH_N * 768 * 2);
  _Float16* h2    = (_Float16*)alloc((size_t)BATCH_N * 128 * 2);
  float*    chat  = (float*)alloc((size_t)BATCH_N * 256 * 4);
  _Float16* g1h   = (_Float16*)alloc((size_t)BATCH_N * 128 * 2);
  float*    gam1  = (float*)alloc((size_t)BATCH_N * 256 * 4);
  _Float16* g2h   = (_Float16*)alloc((size_t)BATCH_N * 128 * 2);
  float*    gam2  = (float*)alloc((size_t)BATCH_N * 256 * 4);
  _Float16* lasth = (_Float16*)alloc((size_t)BATCH_N * 512 * 2);
  _Float16* outh  = (_Float16*)alloc((size_t)BATCH_N * 128 * 2);

  // ---- weight prep ----
  k_pack_lstm_w<<<ew_grid(512 * 448), 256, 0, stream>>>(t_Wih, t_Whh, wT, 512, 300, 128, 448);
  k_pack_lstm_w<<<ew_grid(256 * 160), 256, 0, stream>>>(a_Wih, a_Whh, wA, 256, 81,  64, 160);
  k_pack_lstm_w<<<ew_grid(256 * 448), 256, 0, stream>>>(v_Wih, v_Whh, wV, 256, 371, 64, 448);
  k_convert_f16<<<ew_grid(128 * 512), 256, 0, stream>>>(a1W1, fa1W1, 128 * 512);
  k_convert_f16<<<ew_grid(512 * 128), 256, 0, stream>>>(a1W2, fa1W2, 512 * 128);
  k_convert_f16<<<ew_grid(128 * 512), 256, 0, stream>>>(a2W1, fa2W1, 128 * 512);
  k_convert_f16<<<ew_grid(256 * 128), 256, 0, stream>>>(a2W2, fa2W2, 256 * 128);
  k_convert_f16<<<ew_grid(128 * 768), 256, 0, stream>>>(g1W1, fg1W1, 128 * 768);
  k_convert_f16<<<ew_grid(256 * 128), 256, 0, stream>>>(g1W2, fg1W2, 256 * 128);
  k_convert_f16<<<ew_grid(128 * 768), 256, 0, stream>>>(g2W1, fg2W1, 128 * 768);
  k_convert_f16<<<ew_grid(256 * 128), 256, 0, stream>>>(g2W2, fg2W2, 256 * 128);
  k_convert_f16<<<ew_grid(128 * 512), 256, 0, stream>>>(oW1, foW1, 128 * 512);
  k_convert_f16<<<ew_grid(128), 256, 0, stream>>>(oW2, foW2, 128);

  // ---- state init ----
  hipMemcpyAsync(cT[0], c_t0, (size_t)BATCH_N * 128 * 4, hipMemcpyDeviceToDevice, stream);
  hipMemcpyAsync(cA[0], c_a0, (size_t)BATCH_N * 64 * 4,  hipMemcpyDeviceToDevice, stream);
  hipMemcpyAsync(cV[0], c_v0, (size_t)BATCH_N * 64 * 4,  hipMemcpyDeviceToDevice, stream);
  hipMemcpyAsync(memb, mem0,  (size_t)BATCH_N * 256 * 4, hipMemcpyDeviceToDevice, stream);
  k_zero_f32<<<ew_grid(BATCH_N * 128), 256, 0, stream>>>(hT[0], BATCH_N * 128);
  k_zero_f32<<<ew_grid(BATCH_N * 64),  256, 0, stream>>>(hA[0], BATCH_N * 64);
  k_zero_f32<<<ew_grid(BATCH_N * 64),  256, 0, stream>>>(hV[0], BATCH_N * 64);

  int p = 0, q = 1; // ping-pong index for h/c state
  for (int t = 0; t < SEQ_T; ++t) {
    const float* xt = x_p + (size_t)t * BATCH_N * XP_LD;
    // pack [x|h] operands (zero-padded K)
    k_pack_xh<<<ew_grid(BATCH_N * 448), 256, 0, stream>>>(xt, 0,   300, hT[p], 128, xhT, 448);
    k_pack_xh<<<ew_grid(BATCH_N * 160), 256, 0, stream>>>(xt, 300, 81,  hA[p], 64,  xhA, 160);
    k_pack_xh<<<ew_grid(BATCH_N * 448), 256, 0, stream>>>(xt, 381, 371, hV[p], 64,  xhV, 448);
    // LSTM gate GEMMs (act=none, f32 out)
    gemm_launch<448, 0, true, false>(stream, xhT, 448, wT, 448, t_b, zT, nullptr, 512);
    gemm_launch<160, 0, true, false>(stream, xhA, 160, wA, 160, a_b, zA, nullptr, 256);
    gemm_launch<448, 0, true, false>(stream, xhV, 448, wV, 448, v_b, zV, nullptr, 256);
    // LSTM nonlinearity + state update
    k_lstm_pointwise<<<ew_grid(BATCH_N * 128), 256, 0, stream>>>(zT, cT[p], cT[q], hT[q], 128);
    k_lstm_pointwise<<<ew_grid(BATCH_N * 64),  256, 0, stream>>>(zA, cA[p], cA[q], hA[q], 64);
    k_lstm_pointwise<<<ew_grid(BATCH_N * 64),  256, 0, stream>>>(zV, cV[p], cV[q], hV[q], 64);
    // c_star = [pre_c | cur_c]
    k_pack_cstar<<<ew_grid(BATCH_N * 512), 256, 0, stream>>>(
        cT[p], cA[p], cV[p], cT[q], cA[q], cV[q], cstar);
    // attention MLP1 + softmax*c_star -> both[:,0:512]
    gemm_launch<512, 1, false, true>(stream, cstar, 512, fa1W1, 512, a1b1, nullptr, h1, 128);
    gemm_launch<128, 0, true, false>(stream, h1, 128, fa1W2, 128, a1b2, scores, nullptr, 512);
    k_softmax_attend<<<BATCH_N / 8, 256, 0, stream>>>(scores, cstar, both);
    k_pack_mem<<<ew_grid(BATCH_N * 256), 256, 0, stream>>>(memb, both); // both[:,512:768]
    // c_hat = tanh(MLP2(attended))  -- attended is both[:,0:512], lda=768
    gemm_launch<512, 1, false, true>(stream, both, 768, fa2W1, 512, a2b1, nullptr, h2, 128);
    gemm_launch<128, 3, true, false>(stream, h2, 128, fa2W2, 128, a2b2, chat, nullptr, 256);
    // gamma1/gamma2 = sigmoid(MLP(both))
    gemm_launch<768, 1, false, true>(stream, both, 768, fg1W1, 768, g1b1, nullptr, g1h, 128);
    gemm_launch<128, 2, true, false>(stream, g1h, 128, fg1W2, 128, g1b2, gam1, nullptr, 256);
    gemm_launch<768, 1, false, true>(stream, both, 768, fg2W1, 768, g2b1, nullptr, g2h, 128);
    gemm_launch<128, 2, true, false>(stream, g2h, 128, fg2W2, 128, g2b2, gam2, nullptr, 256);
    // mem = gamma1*mem + gamma2*c_hat
    k_mem_update<<<ew_grid(BATCH_N * 256), 256, 0, stream>>>(memb, gam1, gam2, chat);
    std::swap(p, q);
  }

  // final head: out = MLP([t_h|a_h|v_h|mem])
  k_pack_lasth<<<ew_grid(BATCH_N * 512), 256, 0, stream>>>(hT[p], hA[p], hV[p], memb, lasth);
  gemm_launch<512, 1, false, true>(stream, lasth, 512, foW1, 512, ob1, nullptr, outh, 128);
  k_final<<<ew_grid(BATCH_N), 256, 0, stream>>>(outh, foW2, ob2, (float*)d_out);
}
